// MambaBlock_62208306315887
// MI455X (gfx1250) — matrix operand order repaired and checked by reading, not compiled, not run on hardware
//
#include <hip/hip_runtime.h>
#include <hip/hip_bf16.h>
#include <math.h>

// ---------------------------------------------------------------------------
// Types for CDNA5 WMMA (wave32, v_wmma_f32_16x16x32_bf16)
// ---------------------------------------------------------------------------
typedef __bf16 bf16;
typedef __attribute__((ext_vector_type(8)))  bf16  v8bf;
typedef __attribute__((ext_vector_type(16))) bf16  v16bf;
typedef __attribute__((ext_vector_type(8)))  float v8f;

__device__ __forceinline__ v16bf cat8(v8bf a, v8bf b) {
    return __builtin_shufflevector(a, b, 0, 1, 2, 3, 4, 5, 6, 7,
                                         8, 9, 10, 11, 12, 13, 14, 15);
}

__device__ __forceinline__ float softplusf(float v) {
    if (v > 20.f) return v;
    return log1pf(__expf(v));
}

// ---------------------------------------------------------------------------
// Tiled bf16 WMMA GEMM: C[M,N] = A[M,K] * B[K,N]  (A,B fp32 in memory,
// converted to bf16 in LDS; fp32 accumulate).  Block tile 128x128, 8 waves,
// each wave computes 2x4 16x16 tiles per K=32 step.  Double-buffered LDS,
// register-staged global loads (all 8 b128 loads in flight per tile).
// epi: 0 = plain store, 1 = softplus(acc + bias[n])
// Requires: M % 128 == 0, K % 32 == 0, lda/ldb/ldc % 4 == 0. N arbitrary.
// ---------------------------------------------------------------------------
__global__ __launch_bounds__(256)
void gemm_bf16_wmma(const float* __restrict__ A, const float* __restrict__ Bm,
                    float* __restrict__ C, int M, int N, int K,
                    int lda, int ldb, int ldc,
                    const float* __restrict__ bias, int epi) {
    __shared__ bf16 a_sh [2][128][40];   // [buf][m][k], row stride 80B
    __shared__ bf16 bt_sh[2][128][40];   // [buf][n][k]  (B transposed)

    const int tid   = threadIdx.x;
    const int lane  = tid & 31;
    const int wave  = tid >> 5;
    const int wm    = wave >> 1;   // 0..3 : wave row   (32 rows each)
    const int wn    = wave & 1;    // 0..1 : wave col   (64 cols each)
    const int half  = lane >> 4;   // hi/lo half of wave
    const int l16   = lane & 15;

    const int mbase = blockIdx.y * 128;
    const int nbase = blockIdx.x * 128;

    v8f acc[2][4] = {};
    float4 aR[4], bR[4];

    const int nkt = K >> 5;        // number of K=32 tiles

    // ---- register-staged tile load (no LDS traffic; loads pipeline) ----
    auto loadG = [&](int t) {
        const int kt = t << 5;
        #pragma unroll
        for (int i = 0; i < 4; ++i) {
            int idx = (i * 256 + tid) * 4;          // 0..4095
            int r   = idx >> 5;                     // /32
            int c   = idx & 31;
            aR[i] = *(const float4*)&A[(size_t)(mbase + r) * lda + kt + c];
        }
        #pragma unroll
        for (int i = 0; i < 4; ++i) {
            int idx = (i * 256 + tid) * 4;
            int kk  = idx >> 7;                     // /128
            int nn  = idx & 127;
            int gn  = nbase + nn;
            if (gn + 4 <= N) {
                bR[i] = *(const float4*)&Bm[(size_t)(kt + kk) * ldb + gn];
            } else {
                const float* src = &Bm[(size_t)(kt + kk) * ldb];
                bR[i].x = (gn + 0 < N) ? src[gn + 0] : 0.f;
                bR[i].y = (gn + 1 < N) ? src[gn + 1] : 0.f;
                bR[i].z = (gn + 2 < N) ? src[gn + 2] : 0.f;
                bR[i].w = (gn + 3 < N) ? src[gn + 3] : 0.f;
            }
        }
    };

    // ---- convert fp32 -> bf16 and store registers into LDS buffer ----
    auto storeLDS = [&](int buf) {
        #pragma unroll
        for (int i = 0; i < 4; ++i) {
            int idx = (i * 256 + tid) * 4;
            int r   = idx >> 5;
            int c   = idx & 31;
            a_sh[buf][r][c + 0] = (bf16)aR[i].x;
            a_sh[buf][r][c + 1] = (bf16)aR[i].y;
            a_sh[buf][r][c + 2] = (bf16)aR[i].z;
            a_sh[buf][r][c + 3] = (bf16)aR[i].w;
        }
        #pragma unroll
        for (int i = 0; i < 4; ++i) {
            int idx = (i * 256 + tid) * 4;
            int kk  = idx >> 7;
            int nn  = idx & 127;
            bt_sh[buf][nn + 0][kk] = (bf16)bR[i].x;
            bt_sh[buf][nn + 1][kk] = (bf16)bR[i].y;
            bt_sh[buf][nn + 2][kk] = (bf16)bR[i].z;
            bt_sh[buf][nn + 3][kk] = (bf16)bR[i].w;
        }
    };

    // ---- fragments (layouts per CDNA5 ISA 7.12.2) + 8x WMMA ----
    auto compute = [&](int buf) {
        v16bf af[2];
        #pragma unroll
        for (int i = 0; i < 2; ++i) {
            const bf16* row = &a_sh[buf][wm * 32 + i * 16 + l16][0];
            v8bf lo = *(const v8bf*)(row + half * 8);        // K: half*8 ..
            v8bf hi = *(const v8bf*)(row + 16 + half * 8);   // K: 16+half*8 ..
            af[i] = cat8(lo, hi);
        }
        v16bf bfr[4];
        #pragma unroll
        for (int j = 0; j < 4; ++j) {
            const bf16* row = &bt_sh[buf][wn * 64 + j * 16 + l16][0];
            v8bf lo = *(const v8bf*)(row + half * 8);
            v8bf hi = *(const v8bf*)(row + 16 + half * 8);
            bfr[j] = cat8(lo, hi);
        }
        #pragma unroll
        for (int i = 0; i < 2; ++i)
            #pragma unroll
            for (int j = 0; j < 4; ++j)
                acc[i][j] = __builtin_amdgcn_wmma_f32_16x16x32_bf16(
                    false, af[i], false, bfr[j], (short)0, acc[i][j],
                    false, false);
    };

    // ---- software-pipelined main loop (double-buffered LDS) ----
    loadG(0);
    storeLDS(0);
    for (int t = 0; t < nkt; ++t) {
        const bool has_next = (t + 1 < nkt);
        if (has_next) loadG(t + 1);     // global loads in flight over compute
        __syncthreads();                 // LDS buf (t&1) now valid
        compute(t & 1);
        if (has_next) storeLDS((t + 1) & 1);
    }

    // ---- epilogue + store: VGPR r -> (M = half*8 + r, N = l16) ----
    #pragma unroll
    for (int i = 0; i < 2; ++i) {
        #pragma unroll
        for (int j = 0; j < 4; ++j) {
            int rbase = mbase + wm * 32 + i * 16 + half * 8;
            int ccol  = nbase + wn * 64 + j * 16 + l16;
            if (ccol < N) {
                #pragma unroll
                for (int r = 0; r < 8; ++r) {
                    float v = acc[i][j][r];
                    if (epi == 1) v = softplusf(v + bias[ccol]);
                    C[(size_t)(rbase + r) * ldc + ccol] = v;
                }
            }
        }
    }
}

// ---------------------------------------------------------------------------
// Causal depthwise conv1d (K=4) + bias + SiLU.
// Reads xp half of xz [rows, 2*DI] (cols 0..DI-1), writes xp [rows, DI].
// ---------------------------------------------------------------------------
__global__ void conv_silu_kernel(const float* __restrict__ xz,
                                 const float* __restrict__ conv_w,
                                 const float* __restrict__ conv_b,
                                 float* __restrict__ xp,
                                 int Lc, int DIc) {
    int gid = blockIdx.x * blockDim.x + threadIdx.x;
    int c   = gid % DIc;
    int row = gid / DIc;           // b*L + l
    int l   = row % Lc;
    const int ld = 2 * DIc;
    float acc = conv_b[c];
    #pragma unroll
    for (int j = 0; j < 4; ++j) {
        int li = l + j - 3;
        if (li >= 0)
            acc = fmaf(conv_w[c * 4 + j], xz[(size_t)(row + j - 3) * ld + c], acc);
    }
    xp[gid] = acc / (1.f + __expf(-acc));   // silu
}

// ---------------------------------------------------------------------------
// Selective scan, fused with D-skip and SiLU(z) gate.  One lane per
// (batch, channel): h[16] in registers, 2048 serial steps.  Step t+1's
// operands are prefetched into ping-pong register buffers while step t's
// exp/fma chain executes.  Writes y in place over xp.
// ---------------------------------------------------------------------------
__device__ __forceinline__ void scan_load(const float* __restrict__ xz,
                                          const float* __restrict__ dbl,
                                          const float* __restrict__ dtb,
                                          const float* __restrict__ xp,
                                          size_t row, int c, int DIc, int ldz,
                                          float& dtv, float& xv, float& zv,
                                          float* Bv, float* Cv) {
    dtv = dtb[row * DIc + c];
    xv  = xp [row * DIc + c];
    zv  = xz [row * ldz + DIc + c];
    const float4* b4 = (const float4*)&dbl[row * 96 + 64];  // 16B aligned
    const float4* c4 = (const float4*)&dbl[row * 96 + 80];
    #pragma unroll
    for (int i = 0; i < 4; ++i) {
        float4 b = b4[i], cc = c4[i];
        Bv[4 * i + 0] = b.x;  Bv[4 * i + 1] = b.y;
        Bv[4 * i + 2] = b.z;  Bv[4 * i + 3] = b.w;
        Cv[4 * i + 0] = cc.x; Cv[4 * i + 1] = cc.y;
        Cv[4 * i + 2] = cc.z; Cv[4 * i + 3] = cc.w;
    }
}

__global__ __launch_bounds__(256)
void scan_kernel(const float* __restrict__ xz,   // for z half
                 const float* __restrict__ dbl,  // [rows,96]: dt|B|C
                 const float* __restrict__ dtb,  // [rows,DI] softplus'd dt
                 float* __restrict__ xp,         // in: xp, out: y
                 const float* __restrict__ A_log,
                 const float* __restrict__ Dp,
                 int Lc, int DIc) {
    int cid   = blockIdx.x * blockDim.x + threadIdx.x;  // 0..B*DI-1
    int batch = cid / DIc;
    int c     = cid % DIc;
    const int ldz = 2 * DIc;
    const size_t row0 = (size_t)batch * Lc;

    float Arow[16], h[16];
    #pragma unroll
    for (int s = 0; s < 16; ++s) {
        Arow[s] = -__expf(A_log[c * 16 + s]);
        h[s] = 0.f;
    }
    float Dc = Dp[c];

    float dtv[2], xv[2], zv[2], Bv[2][16], Cv[2][16];
    scan_load(xz, dbl, dtb, xp, row0, c, DIc, ldz,
              dtv[0], xv[0], zv[0], Bv[0], Cv[0]);

    #pragma unroll 2
    for (int t = 0; t < Lc; ++t) {
        const int cur = t & 1, nxt = cur ^ 1;
        if (t + 1 < Lc)
            scan_load(xz, dbl, dtb, xp, row0 + t + 1, c, DIc, ldz,
                      dtv[nxt], xv[nxt], zv[nxt], Bv[nxt], Cv[nxt]);

        float dT = dtv[cur], xT = xv[cur], zT = zv[cur];
        float acc = 0.f;
        #pragma unroll
        for (int s = 0; s < 16; ++s) {
            float dA = __expf(dT * Arow[s]);
            h[s] = fmaf(dA, h[s], (dT * Bv[cur][s]) * xT);
            acc  = fmaf(h[s], Cv[cur][s], acc);
        }
        float y = fmaf(xT, Dc, acc);
        y *= zT / (1.f + __expf(-zT));        // * silu(z)
        xp[(row0 + t) * DIc + c] = y;
    }
}

// ---------------------------------------------------------------------------
// Launch
// ---------------------------------------------------------------------------
extern "C" void kernel_launch(void* const* d_in, const int* in_sizes, int n_in,
                              void* d_out, int out_size, void* d_ws, size_t ws_size,
                              hipStream_t stream) {
    (void)in_sizes; (void)n_in; (void)out_size; (void)ws_size;
    const float* x       = (const float*)d_in[0];
    const float* W_in    = (const float*)d_in[1];
    const float* conv_w  = (const float*)d_in[2];
    const float* conv_b  = (const float*)d_in[3];
    const float* W_xproj = (const float*)d_in[4];
    const float* W_dt    = (const float*)d_in[5];
    const float* b_dt    = (const float*)d_in[6];
    const float* A_log   = (const float*)d_in[7];
    const float* Dp      = (const float*)d_in[8];
    const float* W_out   = (const float*)d_in[9];
    float* out = (float*)d_out;

    const int Bc = 2, Lc = 2048, DIMc = 1024, DIc = 2048;
    const int Mr = Bc * Lc;          // 4096 rows

    float* ws  = (float*)d_ws;
    float* xz  = ws;                                  // [4096, 4096]
    float* xp  = xz  + (size_t)Mr * 2 * DIc;          // [4096, 2048] (becomes y)
    float* dbl = xp  + (size_t)Mr * DIc;              // [4096, 96]
    float* dtb = dbl + (size_t)Mr * 96;               // [4096, 2048]

    dim3 blk(256);

    // 1) xz = x @ W_in                       [4096,1024]x[1024,4096]
    gemm_bf16_wmma<<<dim3(4096 / 128, Mr / 128), blk, 0, stream>>>(
        x, W_in, xz, Mr, 2 * DIc, DIMc, DIMc, 2 * DIc, 2 * DIc, nullptr, 0);

    // 2) xp = silu(depthwise_conv(xp_raw) + conv_b)
    conv_silu_kernel<<<(Mr * DIc) / 256, blk, 0, stream>>>(
        xz, conv_w, conv_b, xp, Lc, DIc);

    // 3) dbl = xp @ W_xproj                  [4096,2048]x[2048,96]
    gemm_bf16_wmma<<<dim3(1, Mr / 128), blk, 0, stream>>>(
        xp, W_xproj, dbl, Mr, 96, DIc, DIc, 96, 96, nullptr, 0);

    // 4) dt = softplus(dbl[:, :64] @ W_dt + b_dt)   [4096,64]x[64,2048]
    gemm_bf16_wmma<<<dim3(DIc / 128, Mr / 128), blk, 0, stream>>>(
        dbl, W_dt, dtb, Mr, DIc, 64, 96, DIc, DIc, b_dt, 1);

    // 5) selective scan + skip + gate (y written over xp)
    scan_kernel<<<(Bc * DIc) / 256, blk, 0, stream>>>(
        xz, dbl, dtb, xp, A_log, Dp, Lc, DIc);

    // 6) out = y @ W_out                     [4096,2048]x[2048,1024]
    gemm_bf16_wmma<<<dim3(DIMc / 128, Mr / 128), blk, 0, stream>>>(
        xp, W_out, out, Mr, DIMc, DIc, DIc, DIMc, DIMc, nullptr, 0);
}
